// LocalMultiHeadAttention_11991548691121
// MI455X (gfx1250) — compile-verified, matrix operand order checked by reading, not run on hardware
//
#include <hip/hip_runtime.h>
#include <stdint.h>

// ---------------------------------------------------------------------------
// LocalMultiHeadAttention for MI455X (gfx1250), bf16 WMMA path.
//   B=2, S=Lk=2048, D=1024, H=16, hd=64, WINDOW=128
// Roofline: dense attn output write = 536 MB (~24us @ 23.3 TB/s) is the
// mandatory-traffic floor.  GEMMs (34 GFLOP) run on v_wmma_f32_16x16x32_bf16
// (f32 accumulate); the multiplicative-mask softmax is restructured so only
// the <=144-wide score window is computed, with all out-of-window columns
// sharing a per-row baseline exp(-m)/Z handled in closed form via V col-sums.
// Global->LDS staging uses GLOBAL_LOAD_ASYNC_TO_LDS_B128 (ASYNCcnt-tracked),
// double-buffered in the GEMM so copies overlap WMMA work.
// ---------------------------------------------------------------------------

typedef __attribute__((ext_vector_type(16))) __bf16 v16bf;
typedef __attribute__((ext_vector_type(8)))  float  v8f;
typedef __attribute__((ext_vector_type(4)))  uint32_t u32x4;
typedef __attribute__((ext_vector_type(4)))  int      i32x4;

#define HD      64
#define S_LEN   2048
#define D_MODEL 1024
#define NH      16
#define BATCH   2
#define NROWS   (BATCH * S_LEN)   // 4096
#define BHTOT   (BATCH * NH)      // 32
#define WIN     128

#if defined(__AMDGCN__) && __has_builtin(__builtin_amdgcn_global_load_async_to_lds_b128) && __has_builtin(__builtin_amdgcn_s_wait_asynccnt)
#define ASYNC_LDS 1
#else
#define ASYNC_LDS 0
#endif

__device__ __forceinline__ __bf16 f2bf(float f) {
  union { float f; uint32_t u; } v; v.f = f;
  uint32_t r = (v.u + 0x7FFFu + ((v.u >> 16) & 1u)) >> 16;
  union { uint16_t s; __bf16 b; } o; o.s = (uint16_t)r; return o.b;
}
__device__ __forceinline__ float bf2f(__bf16 b) {
  union { __bf16 b; uint16_t s; } i; i.b = b;
  union { uint32_t u; float f; } o; o.u = ((uint32_t)i.s) << 16; return o.f;
}

// 16-byte global -> LDS stage (async, ASYNCcnt-tracked, no VGPR round-trip)
__device__ __forceinline__ void stage16(const __bf16* __restrict__ g,
                                        __bf16* __restrict__ l) {
#if ASYNC_LDS
  __builtin_amdgcn_global_load_async_to_lds_b128(
      (__attribute__((address_space(1))) i32x4*)g,
      (__attribute__((address_space(3))) i32x4*)l, 0, 0);
#else
  *(u32x4*)l = *(const u32x4*)g;
#endif
}
__device__ __forceinline__ void stage_wait() {
#if ASYNC_LDS
  __builtin_amdgcn_s_wait_asynccnt(0);
#endif
}
__device__ __forceinline__ void zero16(__bf16* l) {
  u32x4 z = {0u, 0u, 0u, 0u};
  *(u32x4*)l = z;
}

// --------------------------- fp32 -> bf16 conversion -----------------------
__global__ void cvt_f32_bf16(const float* __restrict__ in,
                             __bf16* __restrict__ out, int n) {
  int i = blockIdx.x * blockDim.x + threadIdx.x;
  int stride = gridDim.x * blockDim.x;
  for (; i < n; i += stride) out[i] = f2bf(in[i]);
}

// --------------------------- bf16 GEMM + bias ------------------------------
// C[N x 1024] = A[N x K] * W[K x 1024] + bias.
// WG = 256 thr (8 waves, 2x4 wave grid), per-wave 32x32 (2x2 of 16x16) ->
// WG tile 64x128.  K stepped by 32 through double-buffered LDS with async
// global->LDS staging overlapping the WMMA work.
template <int OUT_BF16>
__global__ __launch_bounds__(256) void gemm_bias(
    const __bf16* __restrict__ A, const __bf16* __restrict__ W,
    const float* __restrict__ bias, void* __restrict__ Cout, int K) {
  __shared__ __bf16 sA[2][64][32];
  __shared__ __bf16 sB[2][32][128];
  const int tid = threadIdx.x;
  const int rowBase = blockIdx.x * 64;
  const int colBase = blockIdx.y * 128;
  const int wave = tid >> 5, lane = tid & 31, hi = lane >> 4, lm = lane & 15;
  const int wr = (wave >> 2) * 32, wc = (wave & 3) * 32;
  v8f acc[2][2] = {};

  auto stage = [&](int buf, int k0) {
    { // A tile: 64x32 halfs = 256 b128 ops, 1 per thread
      int idx = tid * 8;
      int r = idx >> 5, c = idx & 31;
      stage16(&A[(size_t)(rowBase + r) * K + k0 + c], &sA[buf][r][c]);
    }
#pragma unroll
    for (int p = 0; p < 2; ++p) { // B tile: 32x128 halfs = 512 b128 ops
      int idx = tid * 8 + p * 2048;
      int r = idx >> 7, c = idx & 127;
      stage16(&W[(size_t)(k0 + r) * D_MODEL + colBase + c], &sB[buf][r][c]);
    }
  };

  stage(0, 0);
  stage_wait();
  __syncthreads();

  const int steps = K >> 5;
  for (int ks = 0; ks < steps; ++ks) {
    const int buf = ks & 1;
    if (ks + 1 < steps) stage(buf ^ 1, (ks + 1) << 5); // prefetch next K slab

    // 16-bit A 16x32 layout: lane(lm)=row M; halfs[0..7]=K 8*hi+e,
    // halfs[8..15]=K 16+8*hi+e.  B 32x16 mirrors with N=lm.
    v16bf av[2], bv[2];
#pragma unroll
    for (int mr = 0; mr < 2; ++mr)
#pragma unroll
      for (int e = 0; e < 8; ++e) {
        int kk = 8 * hi + e;
        av[mr][e]     = sA[buf][wr + mr * 16 + lm][kk];
        av[mr][e + 8] = sA[buf][wr + mr * 16 + lm][kk + 16];
      }
#pragma unroll
    for (int nr = 0; nr < 2; ++nr)
#pragma unroll
      for (int e = 0; e < 8; ++e) {
        int kk = 8 * hi + e;
        bv[nr][e]     = sB[buf][kk][wc + nr * 16 + lm];
        bv[nr][e + 8] = sB[buf][kk + 16][wc + nr * 16 + lm];
      }
#pragma unroll
    for (int mr = 0; mr < 2; ++mr)
#pragma unroll
      for (int nr = 0; nr < 2; ++nr)
        acc[mr][nr] = __builtin_amdgcn_wmma_f32_16x16x32_bf16(
            false, av[mr], false, bv[nr], (short)0, acc[mr][nr], false, false);

    stage_wait();
    __syncthreads();
  }

#pragma unroll
  for (int mr = 0; mr < 2; ++mr)
#pragma unroll
    for (int nr = 0; nr < 2; ++nr)
#pragma unroll
      for (int r = 0; r < 8; ++r) { // C/D: VGPR r = row r (lanes 0-15) / r+8
        int row = rowBase + wr + mr * 16 + r + 8 * hi;
        int col = colBase + wc + nr * 16 + lm;
        float v = acc[mr][nr][r] + bias[col];
        if (OUT_BF16) ((__bf16*)Cout)[(size_t)row * D_MODEL + col] = f2bf(v);
        else          ((float*)Cout)[(size_t)row * D_MODEL + col]  = v;
      }
}

// --------------------------- V column sum per (b,h) ------------------------
__global__ __launch_bounds__(256) void vsum_kernel(
    const __bf16* __restrict__ V, float* __restrict__ Vsum) {
  __shared__ float red[4][64];
  const int bh = blockIdx.x;
  const int d = threadIdx.x & 63, part = threadIdx.x >> 6;
  const __bf16* base = V + (size_t)bh * (S_LEN * HD);
  float s = 0.f;
  for (int i = part; i < S_LEN; i += 4) s += bf2f(base[(size_t)i * HD + d]);
  red[part][d] = s;
  __syncthreads();
  if (part == 0) Vsum[bh * HD + d] = red[0][d] + red[1][d] + red[2][d] + red[3][d];
}

// --------------------------- windowed attention ----------------------------
// One WG (256 thr, 8 waves) per (bh, 16-query block).  Block key range
// [c0, c0+144) (9 tiles of 16) always covers the union of that block's
// windows.  Out-of-block columns are pure baseline, handled in closed form.
__global__ __launch_bounds__(256) void local_attn(
    const __bf16* __restrict__ Q, const __bf16* __restrict__ Kb,
    const __bf16* __restrict__ Vb, const float* __restrict__ Vsum,
    __bf16* __restrict__ attended, float* __restrict__ attn_out) {
  __shared__ __bf16 sQ[16][64];
  __shared__ __bf16 sK[144][64];
  __shared__ __bf16 sV[160][64];   // rows 144..159 zero (K-pad for P@V)
  __shared__ float  sS[16][160];   // scores, then probs (for attn output)
  __shared__ __bf16 sP[16][160];   // probs (bf16 A-matrix for P@V)
  __shared__ float  sVs[64], sVb[64], sBase[16];

  const int tid = threadIdx.x;
  const int wave = tid >> 5, lane = tid & 31, hi = lane >> 4, lm = lane & 15;
  const int qb = blockIdx.x, bh = blockIdx.y;
  const int i0 = qb * 16;
  const int c0 = (i0 >= 64) ? (i0 - 64) : 0;
  const int nblk = (S_LEN - c0 < 144) ? (S_LEN - c0) : 144;
  const size_t hb = (size_t)bh * (S_LEN * HD);

  if (tid < 128) { // Q tile: 16x64 = 128 b128 ops
    int idx = tid * 8;
    int r = idx >> 6, c = idx & 63;
    stage16(&Q[hb + (size_t)(i0 + r) * HD + c], &sQ[r][c]);
  }
#pragma unroll
  for (int p = 0; p < 5; ++p) { // K tile: 144x64 = 1152 b128 ops
    int op = p * 256 + tid;
    if (op < 1152) {
      int idx = op * 8;
      int t = idx >> 6, c = idx & 63;
      if (t < nblk) stage16(&Kb[hb + (size_t)(c0 + t) * HD + c], &sK[t][c]);
      else          zero16(&sK[t][c]);
    }
  }
#pragma unroll
  for (int p = 0; p < 5; ++p) { // V tile: 160x64 = 1280 b128 ops
    int idx = (p * 256 + tid) * 8;
    int t = idx >> 6, c = idx & 63;
    if (t < nblk) stage16(&Vb[hb + (size_t)(c0 + t) * HD + c], &sV[t][c]);
    else          zero16(&sV[t][c]);
  }
  if (tid < 64) sVs[tid] = Vsum[bh * HD + tid];
  stage_wait();
  __syncthreads();

  // scores = Q * K^T / sqrt(D): wave w -> key tiles w, w+8 (K-loop 2x32)
  const float scale = 0.03125f; // 1/sqrt(1024)
  for (int t = wave; t < 9; t += 8) {
    v8f acc = {};
#pragma unroll
    for (int ks = 0; ks < 2; ++ks) {
      v16bf a, b;
#pragma unroll
      for (int e = 0; e < 8; ++e) {
        int kk = ks * 32 + 8 * hi + e;
        a[e]     = sQ[lm][kk];
        a[e + 8] = sQ[lm][kk + 16];
        b[e]     = sK[t * 16 + lm][kk];
        b[e + 8] = sK[t * 16 + lm][kk + 16];
      }
      acc = __builtin_amdgcn_wmma_f32_16x16x32_bf16(false, a, false, b,
                                                    (short)0, acc, false, false);
    }
#pragma unroll
    for (int r = 0; r < 8; ++r)
      sS[r + 8 * hi][t * 16 + lm] = acc[r] * scale;
  }
  __syncthreads();

  // baseline-aware softmax: masked cols hold score 0 -> m = max(0, max_win s);
  // Z = sum_win exp(s-m) + (Lk - w) * exp(-m)
  if (tid < 16) {
    const int qi = i0 + tid;
    const int wstart = (qi >= 64) ? (qi - 64) : 0;
    int wend = wstart + WIN; if (wend > S_LEN) wend = S_LEN;
    const int a0 = wstart - c0, a1 = wend - c0;
    float m = 0.f;
    for (int c = a0; c < a1; ++c) m = fmaxf(m, sS[tid][c]);
    float z = 0.f;
    for (int c = a0; c < a1; ++c) {
      float e = __expf(sS[tid][c] - m);
      sS[tid][c] = e; z += e;
    }
    const float eb = __expf(-m);
    z += (float)(S_LEN - (wend - wstart)) * eb;
    const float inv = 1.f / z;
    const float base = eb * inv;
    sBase[tid] = base;
    for (int c = 0; c < 160; ++c) {
      float p;
      if (c >= a0 && c < a1)  p = sS[tid][c] * inv; // window prob
      else if (c < nblk)      p = base;             // in-block baseline
      else                    p = 0.f;              // pad / past Lk
      sS[tid][c] = p;
      sP[tid][c] = f2bf(p);
    }
  }
  if (tid >= 64 && tid < 128) { // block V sum (zero-padded rows are safe)
    int d = tid - 64;
    float s = 0.f;
    for (int t = 0; t < 144; ++t) s += bf2f(sV[t][d]);
    sVb[d] = s;
  }
  __syncthreads();

  // attended = P(16x160) @ Vblk(160x64) + base * (Vsum - Vblk_sum)
  if (wave < 4) {
    v8f acc = {};
#pragma unroll
    for (int ks = 0; ks < 5; ++ks) {
      v16bf a, b;
#pragma unroll
      for (int e = 0; e < 8; ++e) {
        int kk = ks * 32 + 8 * hi + e;
        a[e]     = sP[lm][kk];
        a[e + 8] = sP[lm][kk + 16];
        b[e]     = sV[kk][wave * 16 + lm];
        b[e + 8] = sV[kk + 16][wave * 16 + lm];
      }
      acc = __builtin_amdgcn_wmma_f32_16x16x32_bf16(false, a, false, b,
                                                    (short)0, acc, false, false);
    }
#pragma unroll
    for (int r = 0; r < 8; ++r) {
      int i = r + 8 * hi;
      int d = wave * 16 + lm;
      float v = acc[r] + sBase[i] * (sVs[d] - sVb[d]);
      attended[hb + (size_t)(i0 + i) * HD + d] = f2bf(v);
    }
  }

  // dense attn output rows: window probs inside block, baseline elsewhere
  float* arow = attn_out + ((size_t)bh * S_LEN + i0) * S_LEN;
  for (int idx = tid; idx < 16 * S_LEN; idx += 256) {
    int r = idx >> 11, j = idx & (S_LEN - 1);
    int c = j - c0;
    float v = (c >= 0 && c < nblk) ? sS[r][c] : sBase[r];
    arow[(size_t)r * S_LEN + j] = v;
  }
}

// --------------------------- host launcher ---------------------------------
extern "C" void kernel_launch(void* const* d_in, const int* in_sizes, int n_in,
                              void* d_out, int out_size, void* d_ws, size_t ws_size,
                              hipStream_t stream) {
  const float* inputs  = (const float*)d_in[0];
  const float* context = (const float*)d_in[1];
  const float* Wq = (const float*)d_in[2]; const float* bq = (const float*)d_in[3];
  const float* Wk = (const float*)d_in[4]; const float* bk = (const float*)d_in[5];
  const float* Wv = (const float*)d_in[6]; const float* bv = (const float*)d_in[7];
  const float* Wo = (const float*)d_in[8]; const float* bo = (const float*)d_in[9];
  float* out = (float*)d_out;

  const int nAct = NROWS * D_MODEL;       // 4,194,304
  const int nW   = D_MODEL * D_MODEL;     // 1,048,576

  char* ws = (char*)d_ws;
  size_t off = 0;
  auto carve = [&](size_t bytes) -> void* {
    void* p = ws + off;
    off = (off + bytes + 255) & ~(size_t)255;
    return p;
  };
  __bf16* Xq  = (__bf16*)carve((size_t)nAct * 2);
  __bf16* Xc  = (__bf16*)carve((size_t)nAct * 2);
  __bf16* wq  = (__bf16*)carve((size_t)nW * 2);
  __bf16* wk  = (__bf16*)carve((size_t)nW * 2);
  __bf16* wv  = (__bf16*)carve((size_t)nW * 2);
  __bf16* wo  = (__bf16*)carve((size_t)nW * 2);
  __bf16* Qb  = (__bf16*)carve((size_t)nAct * 2);
  __bf16* Kbf = (__bf16*)carve((size_t)nAct * 2);
  __bf16* Vbf = (__bf16*)carve((size_t)nAct * 2);
  __bf16* Att = (__bf16*)carve((size_t)nAct * 2);
  float*  Vsm = (float*)carve((size_t)BHTOT * HD * 4);
  (void)ws_size; (void)in_sizes; (void)n_in; (void)out_size;

  cvt_f32_bf16<<<2048, 256, 0, stream>>>(inputs,  Xq, nAct);
  cvt_f32_bf16<<<2048, 256, 0, stream>>>(context, Xc, nAct);
  cvt_f32_bf16<<<1024, 256, 0, stream>>>(Wq, wq, nW);
  cvt_f32_bf16<<<1024, 256, 0, stream>>>(Wk, wk, nW);
  cvt_f32_bf16<<<1024, 256, 0, stream>>>(Wv, wv, nW);
  cvt_f32_bf16<<<1024, 256, 0, stream>>>(Wo, wo, nW);

  dim3 gg(NROWS / 64, D_MODEL / 128);
  gemm_bias<1><<<gg, 256, 0, stream>>>(Xq, wq, bq, Qb,  D_MODEL);
  gemm_bias<1><<<gg, 256, 0, stream>>>(Xc, wk, bk, Kbf, D_MODEL);
  gemm_bias<1><<<gg, 256, 0, stream>>>(Xc, wv, bv, Vbf, D_MODEL);

  vsum_kernel<<<BHTOT, 256, 0, stream>>>(Vbf, Vsm);

  local_attn<<<dim3(S_LEN / 16, BHTOT), 256, 0, stream>>>(
      Qb, Kbf, Vbf, Vsm, Att, out + (size_t)NROWS * D_MODEL);

  gemm_bias<0><<<gg, 256, 0, stream>>>(Att, wo, bo, out, D_MODEL);
}
